// GRUDCell_3934190043750
// MI455X (gfx1250) — matrix operand order repaired and checked by reading, not compiled, not run on hardware
//
#include <hip/hip_runtime.h>
#include <hip/hip_bf16.h>

// ---------------------------------------------------------------------------
// GRU-D cell for MI455X (gfx1250): bf16 WMMA GEMMs, f32 accumulate.
//   B=16384, F=U=512.  6 GEMMs of 16384x512x512 (~51.5 GFLOP).
// Round-3 changes vs round-2:
//   * global->LDS staging now uses GLOBAL_LOAD_ASYNC_TO_LDS_B128 (ASYNCcnt)
//     via inline asm: no staging VGPRs (kernel back under 256 VGPRs, no
//     s_set_vgpr_msb churn), no ds_store traffic, true DMA-style copy.
//   * pipeline: s_wait_asynccnt 0 -> barrier -> issue next-chunk async loads
//     -> WMMA burst.  One barrier per k-step; async loads have a full 32-WMMA
//     burst to complete before they are waited on.
// Workspace (~116 MiB): see kernel_launch.
// ---------------------------------------------------------------------------

typedef __attribute__((ext_vector_type(16))) __bf16 v16bf;
typedef __attribute__((ext_vector_type(8)))  float  v8f;
typedef __attribute__((ext_vector_type(4)))  float  f4;

#define B_DIM 16384
#define IN_LD 1536          // inputs row stride: [x | m | delta_t]
#define ROWB  80            // LDS bytes per 32-bf16 row (64 data + 16 pad)
#define TILEB (128 * ROWB)  // one 128x32 bf16 tile in LDS (10240 B)

struct Frag32 { f4 lo; f4 hi; };   // 32 bytes == one v16bf fragment

static __device__ inline v8f wmma_bf16(v16bf a, v16bf b, v8f c) {
  // (neg_a, A, neg_b, B, c_mod, C, reuse_a, reuse_b)
  return __builtin_amdgcn_wmma_f32_16x16x32_bf16(
      false, a, false, b, (short)0, c, false, false);
}

// ---- CDNA5 async global->LDS copy (VGLOBAL GLOBAL_LOAD_ASYNC_TO_LDS_B128) --
// vdst = LDS byte address (group-relative; generic-pointer low 32 bits),
// vaddr = 64-bit global address, GV mode.  Tracked by ASYNCcnt.
static __device__ inline void async_ld16(const void* gaddr, unsigned lds_off) {
  asm volatile("global_load_async_to_lds_b128 %0, %1, off"
               :: "v"(lds_off), "v"(gaddr)
               : "memory");
}
static __device__ inline void wait_async0() {
  asm volatile("s_wait_asynccnt 0x0" ::: "memory");
}

// Issue async loads for one 128-row x 32-bf16 tile (512 x 16B chunks,
// 256 threads x 2 chunks; 2 async instructions per wave per tile).
static __device__ inline void g2l_async_tile(char* lds_base, const __bf16* g,
                                             int row0, int k0, int tid) {
#pragma unroll
  for (int i = 0; i < 2; ++i) {
    int idx = tid + 256 * i;
    int r   = idx >> 2;
    int c   = idx & 3;
    async_ld16(g + (size_t)(row0 + r) * 512 + k0 + c * 8,
               (unsigned)(uintptr_t)(lds_base + r * ROWB + c * 16));
  }
}

// A fragment 16x32 bf16 (ISA: lanes 0-15 row=lane K{0..7,16..23};
// lanes 16-31 row=lane-16 K{8..15,24..31}).
static __device__ inline v16bf ld_fragA(const char* ldsA, int row_base, int lane) {
  int sel = lane >> 4;
  int r   = row_base + (lane & 15);
  const f4* p = (const f4*)(ldsA + r * ROWB);
  Frag32 f;
  f.lo = p[sel];
  f.hi = p[2 + sel];
  return __builtin_bit_cast(v16bf, f);
}

// B fragment 32x16 bf16 from N-major LDS tile (lane n holds 32B of K).
static __device__ inline v16bf ld_fragB(const char* ldsB, int n_base, int lane) {
  int sel = lane >> 4;
  int n   = n_base + (lane & 15);
  const f4* p = (const f4*)(ldsB + n * ROWB);
  Frag32 f;
  f.lo = p[sel];
  f.hi = p[2 + sel];
  return __builtin_bit_cast(v16bf, f);
}

// ---------------------------------------------------------------------------
// Kernel 1: elementwise decay.
// ---------------------------------------------------------------------------
__global__ __launch_bounds__(256)
void grud_prep(const float* __restrict__ inputs,
               const float* __restrict__ h_prev,
               const float* __restrict__ gamma_x_decay,
               const float* __restrict__ gamma_h_decay,
               const float* __restrict__ mean_imp,
               __bf16* __restrict__ xd_bf,
               float*  __restrict__ hd_f32,
               __bf16* __restrict__ hd_bf) {
  int idx = blockIdx.x * blockDim.x + threadIdx.x;   // [0, B*512)
  int b = idx >> 9;
  int j = idx & 511;
  const float* row = inputs + (size_t)b * IN_LD;
  float x  = row[j];
  float m  = row[512 + j];
  float dt = row[1024 + j];
  float gx = __expf(-fmaxf(0.0f, gamma_x_decay[j]) * dt);
  float xd = m * x + (1.0f - m) * (gx * x + (1.0f - gx) * mean_imp[j]);
  float gh = __expf(-fmaxf(0.0f, gamma_h_decay[j]) * dt);
  float hd = gh * h_prev[idx];
  xd_bf[idx]  = (__bf16)xd;
  hd_f32[idx] = hd;
  hd_bf[idx]  = (__bf16)hd;
}

// ---------------------------------------------------------------------------
// Kernel 2: transpose + bf16-convert the 6 weight matrices (K-major -> N-major).
// ---------------------------------------------------------------------------
__global__ __launch_bounds__(256)
void grud_wt(const float* __restrict__ W0, const float* __restrict__ W1,
             const float* __restrict__ W2, const float* __restrict__ W3,
             const float* __restrict__ W4, const float* __restrict__ W5,
             __bf16* __restrict__ out) {
  __shared__ float tile[32][33];
  const float* W;
  switch (blockIdx.z) {
    case 0:  W = W0; break;
    case 1:  W = W1; break;
    case 2:  W = W2; break;
    case 3:  W = W3; break;
    case 4:  W = W4; break;
    default: W = W5; break;
  }
  __bf16* O = out + (size_t)blockIdx.z * 512 * 512;
  int k0 = blockIdx.x * 32;
  int n0 = blockIdx.y * 32;
  int tx = threadIdx.x;   // 0..31
  int ty = threadIdx.y;   // 0..7
#pragma unroll
  for (int i = 0; i < 32; i += 8)
    tile[ty + i][tx] = W[(size_t)(k0 + ty + i) * 512 + n0 + tx];
  __syncthreads();
#pragma unroll
  for (int i = 0; i < 32; i += 8)
    O[(size_t)(n0 + ty + i) * 512 + k0 + tx] = (__bf16)tile[tx][ty + i];
}

// ---------------------------------------------------------------------------
// Kernel 3: fused z/r GEMMs + sigmoid.  Block = 128x128 tile, 8 waves in a
// 4x2 grid; each wave computes 32 rows x 64 cols (16 accumulators of v8f).
// Per k-step / wave: 4 A-fragment loads, 16 B-fragment loads, 32 WMMAs,
// 12 async global->LDS instructions for the next chunk.
// ---------------------------------------------------------------------------
__global__ __launch_bounds__(256)
void grud_gemm_zr(const __bf16* __restrict__ xd, const __bf16* __restrict__ hd_bf,
                  const float* __restrict__ hd_f32,
                  const __bf16* __restrict__ wzT, const __bf16* __restrict__ uzT,
                  const __bf16* __restrict__ wrT, const __bf16* __restrict__ urT,
                  const float* __restrict__ b_z, const float* __restrict__ b_r,
                  float* __restrict__ z_out, __bf16* __restrict__ rh_out) {
  __shared__ __align__(16) char lds[2 * 6 * TILEB];   // 120 KB double-buffered
  const int AXD = 0 * TILEB, AHD = 1 * TILEB, BWZ = 2 * TILEB,
            BUZ = 3 * TILEB, BWR = 4 * TILEB, BUR = 5 * TILEB;
  const int BUFSZ = 6 * TILEB;

  int tid    = threadIdx.x;
  int wave   = tid >> 5;
  int lane   = tid & 31;
  int wave_m = wave >> 1;            // 0..3 -> 32-row strip
  int wave_n = wave & 1;             // 0..1 -> 64-col strip
  int n_block = blockIdx.x * 128;
  int m_block = blockIdx.y * 128;

  v8f accZ[8];
  v8f accR[8];
#pragma unroll
  for (int t = 0; t < 8; ++t) { v8f zero = {}; accZ[t] = zero; accR[t] = zero; }

  // Prologue: async-load chunk k0=0 into buffer 0.
  g2l_async_tile(lds + AXD, xd,    m_block, 0, tid);
  g2l_async_tile(lds + AHD, hd_bf, m_block, 0, tid);
  g2l_async_tile(lds + BWZ, wzT,   n_block, 0, tid);
  g2l_async_tile(lds + BUZ, uzT,   n_block, 0, tid);
  g2l_async_tile(lds + BWR, wrT,   n_block, 0, tid);
  g2l_async_tile(lds + BUR, urT,   n_block, 0, tid);

  int cur = 0;
  for (int k0 = 0; k0 < 512; k0 += 32) {
    wait_async0();        // my loads for buffer `cur` have landed in LDS
    __syncthreads();      // everyone's loads landed; everyone done reading nxt

    if (k0 + 32 < 512) {  // DMA next chunk into the other buffer
      char* Ln = lds + (cur ^ 1) * BUFSZ;
      g2l_async_tile(Ln + AXD, xd,    m_block, k0 + 32, tid);
      g2l_async_tile(Ln + AHD, hd_bf, m_block, k0 + 32, tid);
      g2l_async_tile(Ln + BWZ, wzT,   n_block, k0 + 32, tid);
      g2l_async_tile(Ln + BUZ, uzT,   n_block, k0 + 32, tid);
      g2l_async_tile(Ln + BWR, wrT,   n_block, k0 + 32, tid);
      g2l_async_tile(Ln + BUR, urT,   n_block, k0 + 32, tid);
    }

    const char* L = lds + cur * BUFSZ;
    v16bf ax0 = ld_fragA(L + AXD, wave_m * 32,      lane);
    v16bf ax1 = ld_fragA(L + AXD, wave_m * 32 + 16, lane);
    v16bf ah0 = ld_fragA(L + AHD, wave_m * 32,      lane);
    v16bf ah1 = ld_fragA(L + AHD, wave_m * 32 + 16, lane);
#pragma unroll
    for (int t = 0; t < 4; ++t) {
      int nb = wave_n * 64 + t * 16;
      v16bf bwz = ld_fragB(L + BWZ, nb, lane);
      v16bf buz = ld_fragB(L + BUZ, nb, lane);
      v16bf bwr = ld_fragB(L + BWR, nb, lane);
      v16bf bur = ld_fragB(L + BUR, nb, lane);
      accZ[2 * t]     = wmma_bf16(ax0, bwz, accZ[2 * t]);
      accZ[2 * t + 1] = wmma_bf16(ax1, bwz, accZ[2 * t + 1]);
      accZ[2 * t]     = wmma_bf16(ah0, buz, accZ[2 * t]);
      accZ[2 * t + 1] = wmma_bf16(ah1, buz, accZ[2 * t + 1]);
      accR[2 * t]     = wmma_bf16(ax0, bwr, accR[2 * t]);
      accR[2 * t + 1] = wmma_bf16(ax1, bwr, accR[2 * t + 1]);
      accR[2 * t]     = wmma_bf16(ah0, bur, accR[2 * t]);
      accR[2 * t + 1] = wmma_bf16(ah1, bur, accR[2 * t + 1]);
    }
    cur ^= 1;
  }

  // Epilogue.  C/D layout: lanes 0-15 N=lane rows 0..7; lanes 16-31 rows 8..15.
  int lane15 = lane & 15;
  int hi8    = (lane >> 4) * 8;
#pragma unroll
  for (int t = 0; t < 4; ++t) {
    int n = n_block + wave_n * 64 + t * 16 + lane15;
    float bz = b_z[n];
    float br = b_r[n];
#pragma unroll
    for (int mt = 0; mt < 2; ++mt) {
#pragma unroll
      for (int r = 0; r < 8; ++r) {
        int m = m_block + wave_m * 32 + mt * 16 + hi8 + r;
        size_t o = (size_t)m * 512 + n;
        float zv  = 1.0f / (1.0f + __expf(-(accZ[2 * t + mt][r] + bz)));
        float rv  = 1.0f / (1.0f + __expf(-(accR[2 * t + mt][r] + br)));
        float hdv = hd_f32[o];
        z_out[o]  = zv;
        rh_out[o] = (__bf16)(rv * hdv);
      }
    }
  }
}

// ---------------------------------------------------------------------------
// Kernel 4: h_hat GEMM + tanh + final blend into d_out.  Same tiling.
// ---------------------------------------------------------------------------
__global__ __launch_bounds__(256)
void grud_gemm_h(const __bf16* __restrict__ xd, const __bf16* __restrict__ rh,
                 const float* __restrict__ hd_f32, const float* __restrict__ z_in,
                 const __bf16* __restrict__ whT, const __bf16* __restrict__ uhT,
                 const float* __restrict__ b_h,
                 float* __restrict__ h_new) {
  __shared__ __align__(16) char lds[2 * 4 * TILEB];   // 80 KB double-buffered
  const int AXD = 0 * TILEB, ARH = 1 * TILEB, BWH = 2 * TILEB, BUH = 3 * TILEB;
  const int BUFSZ = 4 * TILEB;

  int tid    = threadIdx.x;
  int wave   = tid >> 5;
  int lane   = tid & 31;
  int wave_m = wave >> 1;
  int wave_n = wave & 1;
  int n_block = blockIdx.x * 128;
  int m_block = blockIdx.y * 128;

  v8f accH[8];
#pragma unroll
  for (int t = 0; t < 8; ++t) { v8f zero = {}; accH[t] = zero; }

  g2l_async_tile(lds + AXD, xd,  m_block, 0, tid);
  g2l_async_tile(lds + ARH, rh,  m_block, 0, tid);
  g2l_async_tile(lds + BWH, whT, n_block, 0, tid);
  g2l_async_tile(lds + BUH, uhT, n_block, 0, tid);

  int cur = 0;
  for (int k0 = 0; k0 < 512; k0 += 32) {
    wait_async0();
    __syncthreads();

    if (k0 + 32 < 512) {
      char* Ln = lds + (cur ^ 1) * BUFSZ;
      g2l_async_tile(Ln + AXD, xd,  m_block, k0 + 32, tid);
      g2l_async_tile(Ln + ARH, rh,  m_block, k0 + 32, tid);
      g2l_async_tile(Ln + BWH, whT, n_block, k0 + 32, tid);
      g2l_async_tile(Ln + BUH, uhT, n_block, k0 + 32, tid);
    }

    const char* L = lds + cur * BUFSZ;
    v16bf ax0 = ld_fragA(L + AXD, wave_m * 32,      lane);
    v16bf ax1 = ld_fragA(L + AXD, wave_m * 32 + 16, lane);
    v16bf ar0 = ld_fragA(L + ARH, wave_m * 32,      lane);
    v16bf ar1 = ld_fragA(L + ARH, wave_m * 32 + 16, lane);
#pragma unroll
    for (int t = 0; t < 4; ++t) {
      int nb = wave_n * 64 + t * 16;
      v16bf bwh = ld_fragB(L + BWH, nb, lane);
      v16bf buh = ld_fragB(L + BUH, nb, lane);
      accH[2 * t]     = wmma_bf16(ax0, bwh, accH[2 * t]);
      accH[2 * t + 1] = wmma_bf16(ax1, bwh, accH[2 * t + 1]);
      accH[2 * t]     = wmma_bf16(ar0, buh, accH[2 * t]);
      accH[2 * t + 1] = wmma_bf16(ar1, buh, accH[2 * t + 1]);
    }
    cur ^= 1;
  }

  int lane15 = lane & 15;
  int hi8    = (lane >> 4) * 8;
#pragma unroll
  for (int t = 0; t < 4; ++t) {
    int n = n_block + wave_n * 64 + t * 16 + lane15;
    float bh = b_h[n];
#pragma unroll
    for (int mt = 0; mt < 2; ++mt) {
#pragma unroll
      for (int r = 0; r < 8; ++r) {
        int m = m_block + wave_m * 32 + mt * 16 + hi8 + r;
        size_t o = (size_t)m * 512 + n;
        float hh  = tanhf(accH[2 * t + mt][r] + bh);
        float zv  = z_in[o];
        float hdv = hd_f32[o];
        h_new[o] = (1.0f - zv) * hdv + zv * hh;
      }
    }
  }
}

// ---------------------------------------------------------------------------
extern "C" void kernel_launch(void* const* d_in, const int* in_sizes, int n_in,
                              void* d_out, int out_size, void* d_ws, size_t ws_size,
                              hipStream_t stream) {
  (void)in_sizes; (void)n_in; (void)out_size; (void)ws_size;

  const float* inputs  = (const float*)d_in[0];
  const float* h_prev  = (const float*)d_in[1];
  const float* W_z     = (const float*)d_in[2];
  const float* U_z     = (const float*)d_in[3];
  const float* b_z     = (const float*)d_in[4];
  const float* W_r     = (const float*)d_in[5];
  const float* U_r     = (const float*)d_in[6];
  const float* b_r     = (const float*)d_in[7];
  const float* W_h     = (const float*)d_in[8];
  const float* U_h     = (const float*)d_in[9];
  const float* b_h     = (const float*)d_in[10];
  const float* g_x     = (const float*)d_in[11];
  const float* g_h     = (const float*)d_in[12];
  const float* mean_im = (const float*)d_in[13];
  float* out = (float*)d_out;

  char* ws = (char*)d_ws;
  __bf16* xd_bf  = (__bf16*)(ws);                          // 16 MiB
  float*  hd_f32 = (float*) (ws + ((size_t)16 << 20));     // 32 MiB
  __bf16* hd_bf  = (__bf16*)(ws + ((size_t)48 << 20));     // 16 MiB
  __bf16* wT     = (__bf16*)(ws + ((size_t)64 << 20));     //  3 MiB (6 x 512 KiB)
  float*  z_ws   = (float*) (ws + ((size_t)68 << 20));     // 32 MiB
  __bf16* rh_ws  = (__bf16*)(ws + ((size_t)100 << 20));    // 16 MiB

  const size_t WSTEP = 512 * 512;
  __bf16* wzT = wT + 0 * WSTEP;
  __bf16* uzT = wT + 1 * WSTEP;
  __bf16* wrT = wT + 2 * WSTEP;
  __bf16* urT = wT + 3 * WSTEP;
  __bf16* whT = wT + 4 * WSTEP;
  __bf16* uhT = wT + 5 * WSTEP;

  grud_prep<<<(B_DIM * 512) / 256, 256, 0, stream>>>(
      inputs, h_prev, g_x, g_h, mean_im, xd_bf, hd_f32, hd_bf);

  grud_wt<<<dim3(16, 16, 6), dim3(32, 8), 0, stream>>>(
      W_z, U_z, W_r, U_r, W_h, U_h, wT);

  grud_gemm_zr<<<dim3(512 / 128, B_DIM / 128), 256, 0, stream>>>(
      xd_bf, hd_bf, hd_f32, wzT, uzT, wrT, urT, b_z, b_r, z_ws, rh_ws);

  grud_gemm_h<<<dim3(512 / 128, B_DIM / 128), 256, 0, stream>>>(
      xd_bf, rh_ws, hd_f32, z_ws, whT, uhT, b_h, out);
}
